// LongFormer_79362405695662
// MI455X (gfx1250) — matrix operand order repaired and checked by reading, not compiled, not run on hardware
//
#include <hip/hip_runtime.h>

typedef __attribute__((ext_vector_type(16))) _Float16 v16h;
typedef __attribute__((ext_vector_type(8)))  _Float16 v8h;
typedef __attribute__((ext_vector_type(8)))  float    v8f;
typedef __attribute__((ext_vector_type(4)))  unsigned int v4u;
typedef __attribute__((ext_vector_type(8)))  int      v8i;
typedef __attribute__((ext_vector_type(4)))  int      v4i;

#if defined(__HIP_DEVICE_COMPILE__) && defined(__has_builtin)
#if __has_builtin(__builtin_amdgcn_tensor_load_to_lds) && \
    __has_builtin(__builtin_amdgcn_s_wait_tensorcnt)
#define USE_TDM 1
#endif
#endif

static __device__ __forceinline__ v8f wmma_f16(v16h a, v16h b, v8f c) {
  // D = A(16x32 f16) * B(32x16 f16) + C(16x16 f32)
  return __builtin_amdgcn_wmma_f32_16x16x32_f16(false, a, false, b, (short)0, c,
                                                false, false);
}

// A-fragment (16x32 f16), contiguous k order (B frags use the same):
// lane l=L&15 holds row M=l; hi=L>>4 selects K = 16*hi .. 16*hi+15
static __device__ __forceinline__ v16h load_a_frag(const _Float16* base, int lda,
                                                   int row, int kb, int lane) {
  const int hi = lane >> 4;
  const _Float16* p = base + (size_t)row * lda + kb + 16 * hi;
  v8h lo = *(const v8h*)p;
  v8h hh = *(const v8h*)(p + 8);
  return __builtin_shufflevector(lo, hh, 0, 1, 2, 3, 4, 5, 6, 7, 8, 9, 10, 11,
                                 12, 13, 14, 15);
}

// B-fragment (32x16 f16): lane holds column n = nb + (L&15), K = 16*hi .. +15.
// For out = A*W^T, B^T[n,k] == W[n,k] -> contiguous 32B row segment of W.
static __device__ __forceinline__ v16h load_b_frag(const _Float16* base, int ldb,
                                                   int col, int kb, int lane) {
  const int hi = lane >> 4;
  return *(const v16h*)(base + (size_t)col * ldb + kb + 16 * hi);
}

#ifdef USE_TDM
// Issue one 2D TDM tile load: tile_d1 rows x tile_d0 elems (2-byte elements),
// row pitch stride0 (elements). Rows >= tensor_d1 read as zero (OOB clamp).
// D# bitfields per CDNA5 ISA 8.3/8.4 (group0: count/lds/global/type,
// group1: data_size/dims/tile/strides). Groups 2/3 zero (2D tensor).
static __device__ __forceinline__ void tdm_load_2d(unsigned lds_addr,
                                                   unsigned long long gaddr,
                                                   unsigned tensor_d0,
                                                   unsigned tensor_d1,
                                                   unsigned tile_d0,
                                                   unsigned tile_d1,
                                                   unsigned long long stride0) {
  v4u g0;
  g0[0] = 1u;                                            // count=1, no gather
  g0[1] = lds_addr;                                      // lds_addr [63:32]
  g0[2] = (unsigned)(gaddr & 0xffffffffull);             // global_addr[31:0]
  g0[3] = (unsigned)((gaddr >> 32) & 0x01ffffffull)      // global_addr[56:32]
          | (2u << 30);                                  // type=2 ("image")
  v8i g1;
  g1[0] = (int)(1u << 16);                               // data_size=1 -> 2B
  g1[1] = (int)((tensor_d0 & 0xffffu) << 16);            // td0[15:0] @ [63:48]
  g1[2] = (int)((tensor_d0 >> 16) | ((tensor_d1 & 0xffffu) << 16));
  g1[3] = (int)((tensor_d1 >> 16) | ((tile_d0 & 0xffffu) << 16));
  g1[4] = (int)(tile_d1 & 0xffffu);                      // tile_d1; tile_d2=0
  g1[5] = (int)(stride0 & 0xffffffffull);                // dim0 stride lo
  g1[6] = (int)((stride0 >> 32) & 0xffffull);            // dim0 stride hi
  g1[7] = 0;                                             // dim1 stride (unused)
  v4i z4 = {0, 0, 0, 0};
#if __clang_major__ >= 23
  v8i z8 = {0, 0, 0, 0, 0, 0, 0, 0};
  __builtin_amdgcn_tensor_load_to_lds(g0, g1, z4, z4, z8, 0);
#else
  __builtin_amdgcn_tensor_load_to_lds(g0, g1, z4, z4, 0);
#endif
}
#endif

__global__ void cvt_f32_f16(const float* __restrict__ in, _Float16* __restrict__ out,
                            int n) {
  int i = blockIdx.x * blockDim.x + threadIdx.x;
  if (i < n) out[i] = (_Float16)in[i];
}

// out(MxN) = epilogue(A(MxK) @ W(NxK)^T). One wave computes a 16x64 tile.
template <bool BIAS, bool RELU, bool RESID, bool F16OUT>
__global__ __launch_bounds__(32) void gemm_nt(const _Float16* __restrict__ A,
                                              const _Float16* __restrict__ W,
                                              const float* __restrict__ bias,
                                              const float* __restrict__ resid,
                                              _Float16* __restrict__ outh,
                                              float* __restrict__ outf,
                                              int M, int N, int K) {
  const int lane = threadIdx.x & 31;
  const int l = lane & 15;
  const int hi = lane >> 4;
  const int mb = blockIdx.x * 16;
  const int nb = blockIdx.y * 64;

  v8f acc[4] = {};
  for (int kb = 0; kb < K; kb += 32) {
    v16h a = load_a_frag(A, K, mb + l, kb, lane);
#pragma unroll
    for (int t = 0; t < 4; ++t) {
      v16h b = load_b_frag(W, K, nb + t * 16 + l, kb, lane);
      acc[t] = wmma_f16(a, b, acc[t]);
    }
  }

#pragma unroll
  for (int t = 0; t < 4; ++t) {
    const int col = nb + t * 16 + l;
    float bv = 0.f;
    if constexpr (BIAS) bv = bias[col];
#pragma unroll
    for (int j = 0; j < 8; ++j) {
      const int row = mb + hi * 8 + j;
      float v = acc[t][j] + bv;
      if constexpr (RELU) v = fmaxf(v, 0.f);
      if constexpr (RESID) v += resid[(size_t)row * N + col];
      if constexpr (F16OUT) outh[(size_t)row * N + col] = (_Float16)v;
      else                  outf[(size_t)row * N + col] = v;
    }
  }
}

// Dilated attention, W_WIN=128 DIL=4: query q attends keys q-512+4m, m=0..256,
// zero-padded (OOB key => score exactly 0, still inside softmax).
// Block = 1 wave, 64 queries x 1 head; residue classes r=0..3 give dense
// 16-query x 272-key-grid banded attention (query i -> grid cols [i,i+256]).
// K-grid staged via TDM (tensor_load_to_lds); KV buffer reused for V^T.
__global__ __launch_bounds__(32) void attn_dilated(const _Float16* __restrict__ Q,
                                                   const _Float16* __restrict__ K,
                                                   const _Float16* __restrict__ V,
                                                   _Float16* __restrict__ Xout) {
  constexpr int S = 2048, E = 512;
  // Shared: KV holds K-grid [272][64] during scores, then V^T [64][288] for PV.
  __shared__ __align__(128) _Float16 KV[64 * 288];
  __shared__ __align__(32)  float    Sbuf[16 * 272];
  __shared__ __align__(32)  _Float16 Pbuf[16 * 288];

  const int lane = threadIdx.x & 31;
  const int l = lane & 15;
  const int hi = lane >> 4;
  const int b = blockIdx.x * 64;       // query tile base
  const int fbase = blockIdx.y * 64;   // head feature base
  const v8h zz = {};

  for (int r = 0; r < 4; ++r) {
    const int g0pos = b + r - 512;     // key position of grid column g=0

    // ---- stage K grid [g][feat] into KV ----
    const int gstart = (g0pos < 0) ? ((3 - g0pos) >> 2) : 0;  // first in-range g
    int gend = (S - 1 - g0pos) >> 2;                          // last in-range g
    if (gend > 271) gend = 271;
    const int nvalid = gend - gstart + 1;
#ifdef USE_TDM
    // zero leading OOB rows (left edge blocks only)
    for (int g = lane; g < gstart; g += 32)
#pragma unroll
      for (int f = 0; f < 64; f += 8) *(v8h*)(KV + (size_t)g * 64 + f) = zz;
    {
      const unsigned lds = (unsigned)(size_t)(KV) + (unsigned)gstart * 128u;
      const unsigned long long ga = (unsigned long long)(size_t)(
          K + (size_t)(g0pos + 4 * gstart) * E + fbase);
      // 64-elem rows, pitch 4*E elements; rows >= nvalid auto-zero via OOB.
      tdm_load_2d(lds, ga, 64u, (unsigned)nvalid, 64u, (unsigned)(272 - gstart),
                  (unsigned long long)(4 * E));
    }
    __builtin_amdgcn_s_wait_tensorcnt(0);
#else
    for (int g = lane; g < 272; g += 32) {
      const int p = g0pos + 4 * g;
      _Float16* dst = KV + (size_t)g * 64;
      if (p >= 0 && p < S) {
        const _Float16* src = K + (size_t)p * E + fbase;
#pragma unroll
        for (int f = 0; f < 64; f += 8) *(v8h*)(dst + f) = *(const v8h*)(src + f);
      } else {
#pragma unroll
        for (int f = 0; f < 64; f += 8) *(v8h*)(dst + f) = zz;
      }
    }
#endif
    // zero P (implements band mask + K-padding for PV)
    for (int idx = lane * 8; idx < 16 * 288; idx += 32 * 8)
      *(v8h*)(Pbuf + idx) = zz;
    __syncthreads();

    // ---- scores: S(16x272) = Q_r(16x64) x Kgrid^T ----
    const int qrow = b + r + 4 * l;    // A-fragment row = query (b+r+4*i), i=l
    const v16h qa0 = load_a_frag(Q + fbase, E, qrow, 0, lane);
    const v16h qa1 = load_a_frag(Q + fbase, E, qrow, 32, lane);
    for (int gt = 0; gt < 17; ++gt) {
      const _Float16* kp = KV + (size_t)(gt * 16 + l) * 64 + 16 * hi;
      v8f acc = {};
      acc = wmma_f16(qa0, *(const v16h*)kp, acc);
      acc = wmma_f16(qa1, *(const v16h*)(kp + 32), acc);
#pragma unroll
      for (int j = 0; j < 8; ++j)
        Sbuf[(hi * 8 + j) * 272 + gt * 16 + l] = acc[j];
    }
    __syncthreads();

    // ---- banded softmax over the 257 in-window entries ----
    // reference divides by sqrt(F)=8 AFTER softmax -> fold 0.125 into P
    if (lane < 16) {
      const int i = lane;
      const float* srow = Sbuf + i * 272 + i;
      float mx = -3.0e38f;
      for (int m = 0; m <= 256; ++m) mx = fmaxf(mx, srow[m]);
      float sum = 0.f;
      for (int m = 0; m <= 256; ++m) sum += __expf(srow[m] - mx);
      const float inv = 0.125f / sum;
      _Float16* prow = Pbuf + i * 288 + i;
      for (int m = 0; m <= 256; ++m)
        prow[m] = (_Float16)(__expf(srow[m] - mx) * inv);
    }
    __syncthreads();

    // ---- gather V grid transposed into KV (reused) : Vt[feat][g] ----
    for (int g = lane; g < 288; g += 32) {
      const int p = g0pos + 4 * g;
      if (g < 272 && p >= 0 && p < S) {
        const _Float16* vp = V + (size_t)p * E + fbase;
        for (int f = 0; f < 64; ++f) KV[f * 288 + g] = vp[f];
      } else {
        for (int f = 0; f < 64; ++f) KV[f * 288 + g] = (_Float16)0.f;
      }
    }
    __syncthreads();

    // ---- O(16x64) = P(16x288) x Vgrid(288x64), A frag reused across 4 tiles
    v8f acc[4] = {};
    for (int kb = 0; kb < 288; kb += 32) {
      v16h a = load_a_frag(Pbuf, 288, l, kb, lane);
#pragma unroll
      for (int t = 0; t < 4; ++t) {
        v16h bf = *(const v16h*)(KV + (size_t)(t * 16 + l) * 288 + kb + 16 * hi);
        acc[t] = wmma_f16(a, bf, acc[t]);
      }
    }
#pragma unroll
    for (int t = 0; t < 4; ++t) {
#pragma unroll
      for (int j = 0; j < 8; ++j) {
        const int q = b + r + 4 * (hi * 8 + j);
        Xout[(size_t)q * E + fbase + t * 16 + l] = (_Float16)acc[t][j];
      }
    }
    __syncthreads();
  }
}

extern "C" void kernel_launch(void* const* d_in, const int* in_sizes, int n_in,
                              void* d_out, int out_size, void* d_ws, size_t ws_size,
                              hipStream_t stream) {
  (void)in_sizes; (void)n_in; (void)out_size; (void)ws_size;
  const float* x  = (const float*)d_in[0];
  const float* Wq = (const float*)d_in[1];
  const float* Wk = (const float*)d_in[2];
  const float* Wv = (const float*)d_in[3];
  const float* W1 = (const float*)d_in[4];
  const float* b1 = (const float*)d_in[5];
  const float* W2 = (const float*)d_in[6];
  const float* b2 = (const float*)d_in[7];
  float* out = (float*)d_out;

  const int S = 2048, E = 512;
  const size_t NX = (size_t)S * E;   // 1,048,576
  const size_t NW = (size_t)E * E;   //   262,144

  _Float16* p = (_Float16*)d_ws;
  _Float16* xh  = p; p += NX;
  _Float16* Wqh = p; p += NW;
  _Float16* Wkh = p; p += NW;
  _Float16* Wvh = p; p += NW;
  _Float16* W1h = p; p += NW;
  _Float16* W2h = p; p += NW;
  _Float16* Qh  = p; p += NX;
  _Float16* Kh  = p; p += NX;
  _Float16* Vh  = p; p += NX;
  _Float16* X1h = p; p += NX;
  _Float16* Fh  = p; p += NX;

  // f32 -> f16 conversions
  cvt_f32_f16<<<dim3((unsigned)((NX + 255) / 256)), dim3(256), 0, stream>>>(x, xh, (int)NX);
  cvt_f32_f16<<<dim3((unsigned)((NW + 255) / 256)), dim3(256), 0, stream>>>(Wq, Wqh, (int)NW);
  cvt_f32_f16<<<dim3((unsigned)((NW + 255) / 256)), dim3(256), 0, stream>>>(Wk, Wkh, (int)NW);
  cvt_f32_f16<<<dim3((unsigned)((NW + 255) / 256)), dim3(256), 0, stream>>>(Wv, Wvh, (int)NW);
  cvt_f32_f16<<<dim3((unsigned)((NW + 255) / 256)), dim3(256), 0, stream>>>(W1, W1h, (int)NW);
  cvt_f32_f16<<<dim3((unsigned)((NW + 255) / 256)), dim3(256), 0, stream>>>(W2, W2h, (int)NW);

  const dim3 gg(S / 16, E / 64), bb(32);
  // Q/K/V projections: out = x @ W^T
  gemm_nt<false, false, false, true><<<gg, bb, 0, stream>>>(xh, Wqh, nullptr, nullptr, Qh, nullptr, S, E, E);
  gemm_nt<false, false, false, true><<<gg, bb, 0, stream>>>(xh, Wkh, nullptr, nullptr, Kh, nullptr, S, E, E);
  gemm_nt<false, false, false, true><<<gg, bb, 0, stream>>>(xh, Wvh, nullptr, nullptr, Vh, nullptr, S, E, E);

  // Dilated attention: 32 query tiles x 8 heads
  attn_dilated<<<dim3(S / 64, 8), dim3(32), 0, stream>>>(Qh, Kh, Vh, X1h);

  // FFN: f = relu(x1 @ W1^T + b1); out = x + relu(f @ W2^T + b2)
  gemm_nt<true, true, false, true><<<gg, bb, 0, stream>>>(X1h, W1h, b1, nullptr, Fh, nullptr, S, E, E);
  gemm_nt<true, true, true, false><<<gg, bb, 0, stream>>>(Fh, W2h, b2, x, nullptr, out, S, E, E);
}
